// SGCNMPModule_68152541053498
// MI455X (gfx1250) — matrix operand order repaired and checked by reading, not compiled, not run on hardware
//
#include <hip/hip_runtime.h>
#include <hip/hip_bf16.h>

typedef __attribute__((ext_vector_type(16))) __bf16 v16bf;
typedef __attribute__((ext_vector_type(8)))  float  v8f;
typedef unsigned short u16;

#define DFEAT 128
#define EPS 1e-5f

__device__ __forceinline__ u16 f2bf(float f) {
    unsigned u = __float_as_uint(f);
    unsigned r = u + 0x7FFFu + ((u >> 16) & 1u);   // round-to-nearest-even
    return (u16)(r >> 16);
}

__device__ __forceinline__ void atomAddF(float* p, float v) {
    unsafeAtomicAdd(p, v);   // lowers to global_atomic_add_f32
}

// ---------------- small utility kernels ----------------

__global__ void k_zero(float* __restrict__ p, int n) {
    int i = blockIdx.x * blockDim.x + threadIdx.x;
    if (i < n) p[i] = 0.0f;
}

__global__ void k_deg(const long long* __restrict__ ei, const float* __restrict__ ew,
                      float* __restrict__ deg, int E) {
    int e = blockIdx.x * blockDim.x + threadIdx.x;
    if (e >= E) return;
    long long dst = ei[(size_t)E + e];
    atomAddF(deg + dst, ew[e]);
}

__global__ void k_dinv(const float* __restrict__ deg, float* __restrict__ dinv, int n) {
    int i = blockIdx.x * blockDim.x + threadIdx.x;
    if (i >= n) return;
    float d = deg[i];
    dinv[i] = (d > 0.0f) ? rsqrtf(d) : 0.0f;
}

__global__ void k_norm(const long long* __restrict__ ei, const float* __restrict__ ew,
                       const float* __restrict__ dinv, float* __restrict__ nrm, int E) {
    int e = blockIdx.x * blockDim.x + threadIdx.x;
    if (e >= E) return;
    long long s = ei[e];
    long long d = ei[(size_t)E + e];
    nrm[e] = dinv[s] * ew[e] * dinv[d];
}

__global__ void k_f32_to_bf16(const float* __restrict__ src, u16* __restrict__ dst, int n) {
    int i = blockIdx.x * blockDim.x + threadIdx.x;
    if (i < n) dst[i] = f2bf(src[i]);
}

// Wt[j*128 + k] = bf16(W[k*128 + j])  (transpose so B loads are contiguous in k)
__global__ void k_transposeW(const float* __restrict__ W, u16* __restrict__ Wt) {
    int i = blockIdx.x * blockDim.x + threadIdx.x;   // 0 .. 16383
    if (i >= DFEAT * DFEAT) return;
    int j = i >> 7, k = i & 127;
    Wt[j * DFEAT + k] = f2bf(W[k * DFEAT + j]);
}

// ---------------- WMMA GEMM: xl = h(bf16) @ W -> f32 ----------------
// One wave computes one 16x16 tile; 8 waves per block cover all 8 col tiles
// of a 16-row strip. K = 128 via 4x v_wmma_f32_16x16x32_bf16.
__global__ void k_gemm_wmma(const u16* __restrict__ A,   // [rows x 128] bf16 row-major
                            const u16* __restrict__ Bt,  // [128 x 128] bf16, Bt[col][k]
                            float* __restrict__ C,       // [rows x 128] f32
                            int rows) {
    int lane = threadIdx.x & 31;
    int wave = threadIdx.x >> 5;        // 0..7 -> column tile
    int rowBase = blockIdx.x * 16;
    if (rowBase + 16 > rows) return;    // uniform per block
    int colBase = wave * 16;
    int m  = lane & 15;
    int hi = lane >> 4;

    const u16* arow = A  + (size_t)(rowBase + m) * DFEAT;
    const u16* bcol = Bt + (size_t)(colBase + m) * DFEAT;

    v8f acc = {};
    union { v16bf v; uint4 q[2]; } a, b;
#pragma unroll
    for (int k0 = 0; k0 < DFEAT; k0 += 32) {
        // A 16x32 layout: lanes 0-15 row m hold K = k0+{0..7, 16..23}; lanes 16-31 +8
        a.q[0] = *(const uint4*)(arow + k0 + hi * 8);
        a.q[1] = *(const uint4*)(arow + k0 + 16 + hi * 8);
        // B 32x16 layout: lanes 0-15 col m hold K = k0+{0..7, 16..23}; lanes 16-31 +8
        b.q[0] = *(const uint4*)(bcol + k0 + hi * 8);
        b.q[1] = *(const uint4*)(bcol + k0 + 16 + hi * 8);
        acc = __builtin_amdgcn_wmma_f32_16x16x32_bf16(false, a.v, false, b.v,
                                                      (short)0, acc, false, false);
    }
    // C/D layout: VGPR r -> M = r (lanes 0-15) / M = 8+r (lanes 16-31), N = m
    float* crow = C + (size_t)(rowBase + hi * 8) * DFEAT + colBase + m;
#pragma unroll
    for (int r = 0; r < 8; ++r) crow[(size_t)r * DFEAT] = acc[r];
}

// ---------------- message passing ----------------

__global__ void k_init_agg(const float* __restrict__ bias, float* __restrict__ agg, int n) {
    int i = blockIdx.x * blockDim.x + threadIdx.x;
    if (i < n) agg[i] = bias[i & 127];
}

// one thread = one (edge, 4-feature chunk); 32 consecutive threads share an edge
__global__ void k_edges(const long long* __restrict__ ei, const float* __restrict__ nrm,
                        const float* __restrict__ xl, float* __restrict__ agg, int E) {
    int gid = blockIdx.x * blockDim.x + threadIdx.x;
    int e = gid >> 5;
    if (e >= E) return;
    int c = gid & 31;
    long long s = ei[e];
    long long d = ei[(size_t)E + e];
    float w = nrm[e];
    float4 v = ((const float4*)(xl + (size_t)s * DFEAT))[c];
    float* ap = agg + (size_t)d * DFEAT + c * 4;
    atomAddF(ap + 0, w * v.x);
    atomAddF(ap + 1, w * v.y);
    atomAddF(ap + 2, w * v.z);
    atomAddF(ap + 3, w * v.w);
}

// hlast = agg (+ hlast if not first layer)
__global__ void k_residual(const float* __restrict__ agg, float* __restrict__ hlast,
                           int n, int first) {
    int i = blockIdx.x * blockDim.x + threadIdx.x;
    if (i >= n) return;
    float v = agg[i];
    if (!first) v += hlast[i];
    hlast[i] = v;
}

// ---------------- batch norm ----------------

__global__ void k_stats(const float* __restrict__ hr, float* __restrict__ stats, int rows) {
    int c = threadIdx.x;                 // 0..127, one column per thread
    float s = 0.0f, ss = 0.0f;
    for (int r = blockIdx.x; r < rows; r += gridDim.x) {
        float v = hr[(size_t)r * DFEAT + c];
        s += v;
        ss += v * v;
    }
    atomAddF(stats + c, s);
    atomAddF(stats + DFEAT + c, ss);
}

__global__ void k_apply(const float* __restrict__ hr, const float* __restrict__ stats,
                        const float* __restrict__ g, const float* __restrict__ bt,
                        u16* __restrict__ hbf, float* __restrict__ outF,
                        int n, float invN) {
    int i = blockIdx.x * blockDim.x + threadIdx.x;
    if (i >= n) return;
    int c = i & 127;
    float mean = stats[c] * invN;
    float var  = stats[DFEAT + c] * invN - mean * mean;
    float rs   = rsqrtf(var + EPS);
    float v = (hr[i] - mean) * rs * g[c] + bt[c];
    v = v > 0.0f ? v : 0.0f;
    hbf[i]  = f2bf(v);
    outF[i] = v;
}

// ---------------- launcher ----------------

extern "C" void kernel_launch(void* const* d_in, const int* in_sizes, int n_in,
                              void* d_out, int out_size, void* d_ws, size_t ws_size,
                              hipStream_t stream) {
    const float*     x     = (const float*)d_in[0];
    const long long* ei    = (const long long*)d_in[1];
    const float*     ew    = (const float*)d_in[2];
    const float*     W     = (const float*)d_in[3];
    const float*     bias  = (const float*)d_in[4];
    const float*     gamma = (const float*)d_in[5];
    const float*     beta  = (const float*)d_in[6];

    const int N = in_sizes[0] / DFEAT;
    const int E = in_sizes[2];
    const int L = in_sizes[3] / (DFEAT * DFEAT);
    const int ND = N * DFEAT;

    // workspace layout (all 16B-aligned)
    char* p = (char*)d_ws;
    float* deg   = (float*)p;            p += (size_t)N * 4;
    float* dinv  = (float*)p;            p += (size_t)N * 4;
    float* nrm   = (float*)p;            p += (size_t)E * 4;
    float* xl    = (float*)p;            p += (size_t)ND * 4;
    float* agg   = (float*)p;            p += (size_t)ND * 4;
    float* hlast = (float*)p;            p += (size_t)ND * 4;
    float* stats = (float*)p;            p += 256 * 4;
    u16*   hbf   = (u16*)p;              p += (size_t)ND * 2;
    u16*   wt    = (u16*)p;              p += (size_t)DFEAT * DFEAT * 2;

    const int TB = 256;
    const int gN   = (N + TB - 1) / TB;
    const int gE   = (E + TB - 1) / TB;
    const int gND  = (ND + TB - 1) / TB;
    const int gEC  = (E * 32 + TB - 1) / TB;   // edge * feature-chunk threads
    const int gWW  = (DFEAT * DFEAT + TB - 1) / TB;
    const int gGemm = N / 16;                  // 16-row strips (N % 16 == 0)

    // degree + normalization
    k_zero<<<gN, TB, 0, stream>>>(deg, N);
    k_deg<<<gE, TB, 0, stream>>>(ei, ew, deg, E);
    k_dinv<<<gN, TB, 0, stream>>>(deg, dinv, N);
    k_norm<<<gE, TB, 0, stream>>>(ei, ew, dinv, nrm, E);

    // h0 = x in bf16
    k_f32_to_bf16<<<gND, TB, 0, stream>>>(x, hbf, ND);

    const float invN = 1.0f / (float)N;
    for (int l = 0; l < L; ++l) {
        k_transposeW<<<gWW, TB, 0, stream>>>(W + (size_t)l * DFEAT * DFEAT, wt);
        k_gemm_wmma<<<gGemm, 256, 0, stream>>>(hbf, wt, xl, N);
        k_init_agg<<<gND, TB, 0, stream>>>(bias + (size_t)l * DFEAT, agg, ND);
        k_edges<<<gEC, TB, 0, stream>>>(ei, nrm, xl, agg, E);
        k_residual<<<gND, TB, 0, stream>>>(agg, hlast, ND, l == 0 ? 1 : 0);
        k_zero<<<1, 256, 0, stream>>>(stats, 256);
        k_stats<<<512, DFEAT, 0, stream>>>(hlast, stats, N);
        float* outF = (l == L - 1) ? (float*)d_out : xl;   // xl is dead here for l<L-1
        k_apply<<<gND, TB, 0, stream>>>(hlast, stats,
                                        gamma + (size_t)l * DFEAT,
                                        beta + (size_t)l * DFEAT,
                                        hbf, outF, ND, invN);
    }
}
